// DecoupledPointJAFAR_28819230556939
// MI455X (gfx1250) — matrix operand order repaired and checked by reading, not compiled, not run on hardware
//
#include <hip/hip_runtime.h>
#include <hip/hip_bf16.h>

typedef __attribute__((ext_vector_type(16))) _Float16 v16h;
typedef __attribute__((ext_vector_type(8)))  float    v8f;

#define WMMA_F16(c_, a_, b_) \
  (c_) = __builtin_amdgcn_wmma_f32_16x16x32_f16(false, (a_), false, (b_), (short)0, (c_), false, false)

// ---------------- problem sizes ----------------
constexpr int BATCH = 4, NPTS = 8192, BN = BATCH * NPTS;
constexpr int QKD = 64, KNNK = 16, GEO = 12, SEMD = 192, NCLS = 13;

// ---------------- d_out layout (floats, tuple concat order) ----------------
constexpr size_t O_LOGITS = 0;                              // (BN,13)
constexpr size_t O_AFF    = O_LOGITS + (size_t)BN * NCLS;   // (BN,16)
constexpr size_t O_KIDX   = O_AFF    + (size_t)BN * KNNK;   // (BN,16)
constexpr size_t O_REF    = O_KIDX   + (size_t)BN * KNNK;   // (BN,64)
constexpr size_t O_BDY    = O_REF    + (size_t)BN * QKD;    // (BN,)

// ---------------- workspace layout (bytes) ----------------
constexpr size_t WS_KIDX  = 0;                               // int[BN*16]
constexpr size_t WS_Q     = WS_KIDX + (size_t)BN * KNNK * 4; // float[BN*64]
constexpr size_t WS_K     = WS_Q   + (size_t)BN * QKD * 4;
constexpr size_t WS_V     = WS_K   + (size_t)BN * QKD * 4;
constexpr size_t WS_PARAM = WS_V   + (size_t)BN * QKD * 4;   // multiple of 1024

// param block: weights pre-swizzled into WMMA B-fragment layout.
// One fragment = 32 lanes x 16 halfs = 1024 bytes.  f = ct*nkt + kt.
constexpr int PW_GE1 = 0;                    // 4 ct x 1 kt  = 4 frags
constexpr int PW_GE2 = PW_GE1 + 4  * 1024;   // 4 x 2 = 8
constexpr int PW_VP  = PW_GE2 + 8  * 1024;   // 4 x 6 = 24
constexpr int PW_QW  = PW_VP  + 24 * 1024;   // 8
constexpr int PW_KW  = PW_QW  + 8  * 1024;   // 8
constexpr int PW_RP1 = PW_KW  + 8  * 1024;   // 4
constexpr int PW_RP2 = PW_RP1 + 4  * 1024;   // 8
constexpr int PW_BD1 = PW_RP2 + 8  * 1024;   // 2 x 2 = 4
constexpr int PB_GE1B = PW_BD1 + 4 * 1024;   // folded biases (f32)
constexpr int PB_GE2B = PB_GE1B + 64 * 4;
constexpr int PB_VPB  = PB_GE2B + 64 * 4;
constexpr int PB_QB   = PB_VPB  + 64 * 4;
constexpr int PB_KB   = PB_QB   + 64 * 4;
constexpr int PB_RP1B = PB_KB   + 64 * 4;
constexpr int PB_RP2B = PB_RP1B + 64 * 4;
constexpr int PB_BD1B = PB_RP2B + 64 * 4;
constexpr int PB_BD2W = PB_BD1B + 32 * 4;
constexpr int PB_BD2B = PB_BD2W + 32 * 4;

// ---------------- A-fragment builder (wave32, CDNA5 §7.12.2 layout) ----------------
// Unguarded: caller guarantees every generated k index is valid/initialized.
// A 16x32 f16 from row-major f32 rows. lane<16: M=lane, K in {0-7,16-23}; lanes>=16: +8.
__device__ __forceinline__ v16h a_frag_rows(const float* base, int stride, int lane, int ko) {
  const int m   = lane & 15;
  const int hib = (lane & 16) ? 8 : 0;
  const float* row = base + m * stride + ko + hib;
  v16h a;
#pragma unroll
  for (int rr = 0; rr < 8; ++rr) {
    const int kk = ((rr < 4) ? 0 : 16) + ((rr & 3) << 1);
    const float2 w = *(const float2*)(row + kk);   // 8B-aligned pair (stride/ko/hib all even)
    a[2 * rr]     = (_Float16)w.x;
    a[2 * rr + 1] = (_Float16)w.y;
  }
  return a;
}

// ---------------- kernel 0: fold BN, pre-swizzle weights into B-fragment layout ----------------
__device__ __forceinline__ void pack_wfrag(_Float16* dst, const float* W, const float* g,
                                           float rs, int cout, int cin, int nct, int nkt,
                                           int tid, int nthreads) {
  const int total = nct * nkt * 32 * 16;
  for (int i = tid; i < total; i += nthreads) {
    const int e    = i & 15;
    const int lane = (i >> 4) & 31;
    const int f    = i >> 9;
    const int kt   = f % nkt, ct = f / nkt;
    const int n = ct * 16 + (lane & 15);
    const int k = kt * 32 + ((lane & 16) ? 8 : 0) + (e < 8 ? e : e + 8);
    float v = 0.0f;
    if (k < cin && n < cout) {
      const float s = g ? (g[n] * rs) : 1.0f;
      v = W[n * cin + k] * s;
    }
    dst[i] = (_Float16)v;
  }
}

__global__ __launch_bounds__(256) void prep_kernel(
    const float* ge_w1, const float* ge_b1, const float* ge_g1, const float* ge_be1,
    const float* ge_w2, const float* ge_b2, const float* ge_g2, const float* ge_be2,
    const float* vp_w,  const float* vp_b,  const float* vp_g,  const float* vp_be,
    const float* q_w,   const float* q_b,   const float* k_w,   const float* k_b,
    const float* rp_w1, const float* rp_b1, const float* rp_g1, const float* rp_be1,
    const float* rp_w2, const float* rp_b2,
    const float* bd_w1, const float* bd_b1, const float* bd_g1, const float* bd_be1,
    const float* bd_w2, const float* bd_b2,
    unsigned char* wb) {
  const int t = threadIdx.x;
  const float rs = rsqrtf(1.0f + 1e-5f);

  pack_wfrag((_Float16*)(wb + PW_GE1), ge_w1, ge_g1, rs, 64, GEO, 4, 1, t, 256);
  pack_wfrag((_Float16*)(wb + PW_GE2), ge_w2, ge_g2, rs, 64, 64, 4, 2, t, 256);
  pack_wfrag((_Float16*)(wb + PW_VP),  vp_w,  vp_g,  rs, 64, SEMD, 4, 6, t, 256);
  pack_wfrag((_Float16*)(wb + PW_QW),  q_w,   nullptr, 1.0f, 64, 64, 4, 2, t, 256);
  pack_wfrag((_Float16*)(wb + PW_KW),  k_w,   nullptr, 1.0f, 64, 64, 4, 2, t, 256);
  pack_wfrag((_Float16*)(wb + PW_RP1), rp_w1, rp_g1, rs, 64, 3, 4, 1, t, 256);
  pack_wfrag((_Float16*)(wb + PW_RP2), rp_w2, nullptr, 1.0f, 64, 64, 4, 2, t, 256);
  pack_wfrag((_Float16*)(wb + PW_BD1), bd_w1, bd_g1, rs, 32, 64, 2, 2, t, 256);

  float* ge1b = (float*)(wb + PB_GE1B);
  float* ge2b = (float*)(wb + PB_GE2B);
  float* vpb  = (float*)(wb + PB_VPB);
  float* qb   = (float*)(wb + PB_QB);
  float* kb   = (float*)(wb + PB_KB);
  float* r1b  = (float*)(wb + PB_RP1B);
  float* r2b  = (float*)(wb + PB_RP2B);
  float* b1b  = (float*)(wb + PB_BD1B);
  float* b2w  = (float*)(wb + PB_BD2W);
  float* b2b  = (float*)(wb + PB_BD2B);
  for (int o = t; o < 64; o += 256) {
    ge1b[o] = ge_b1[o] * (ge_g1[o] * rs) + ge_be1[o];
    ge2b[o] = ge_b2[o] * (ge_g2[o] * rs) + ge_be2[o];
    vpb[o]  = vp_b[o]  * (vp_g[o]  * rs) + vp_be[o];
    qb[o]   = q_b[o];
    kb[o]   = k_b[o];
    r1b[o]  = rp_b1[o] * (rp_g1[o] * rs) + rp_be1[o];
    r2b[o]  = rp_b2[o];
  }
  for (int o = t; o < 32; o += 256) {
    b1b[o] = bd_b1[o] * (bd_g1[o] * rs) + bd_be1[o];
    b2w[o] = bd_w2[o];
  }
  if (t == 0) b2b[0] = bd_b2[0];
}

// ---------------- kernel 1: brute-force KNN, register-resident top-16 ----------------
__global__ __launch_bounds__(256) void knn_kernel(const float* __restrict__ xyz,
                                                  int* __restrict__ kidx_ws,
                                                  float* __restrict__ kidx_out) {
  __shared__ float sx[256], sy[256], sz[256];
  const int gid = blockIdx.x * 256 + threadIdx.x;   // blocks never straddle batches
  const int b = gid >> 13;
  const int n = gid & (NPTS - 1);
  const float* P = xyz + (size_t)b * NPTS * 3;
  const float qx = P[n * 3 + 0], qy = P[n * 3 + 1], qz = P[n * 3 + 2];

  float dist[KNNK];
  int   idx[KNNK];
#pragma unroll
  for (int i = 0; i < KNNK; ++i) { dist[i] = 3.4e38f; idx[i] = 0; }

  for (int c0 = 0; c0 < NPTS; c0 += 256) {
    __syncthreads();
    const int t = threadIdx.x;
    sx[t] = P[(c0 + t) * 3 + 0];
    sy[t] = P[(c0 + t) * 3 + 1];
    sz[t] = P[(c0 + t) * 3 + 2];
    if (c0 + 256 < NPTS) __builtin_prefetch(&P[(c0 + 256 + t) * 3], 0, 1);
    __syncthreads();
    for (int t2 = 0; t2 < 256; ++t2) {
      const float dx = qx - sx[t2], dy = qy - sy[t2], dz = qz - sz[t2];
      const float d = dx * dx + dy * dy + dz * dz;
      if (d < dist[KNNK - 1]) {                 // rare after warm-up
        const int jj = c0 + t2;
#pragma unroll
        for (int i = KNNK - 1; i > 0; --i) {    // fully unrolled -> v_cndmask, stays in VGPRs
          if (d < dist[i]) {
            if (d < dist[i - 1]) { dist[i] = dist[i - 1]; idx[i] = idx[i - 1]; }
            else                 { dist[i] = d;           idx[i] = jj; }
          }
        }
        if (d < dist[0]) { dist[0] = d; idx[0] = jj; }
      }
    }
  }
#pragma unroll
  for (int i = 0; i < KNNK; ++i) {
    kidx_ws[gid * KNNK + i]  = idx[i];
    kidx_out[gid * KNNK + i] = (float)idx[i];
  }
}

// ---------------- kernel 2: per-point feature pipeline (all WMMA) ----------------
__global__ __launch_bounds__(128) void feat_kernel(
    const float* __restrict__ jf, const float* __restrict__ sem,
    const unsigned char* __restrict__ wb,
    float* __restrict__ Qo, float* __restrict__ Ko, float* __restrict__ Vo,
    float* __restrict__ bdy_out) {
  __shared__ float Xs[4][16 * 64];
  const int lane = threadIdx.x & 31;
  const int wv   = threadIdx.x >> 5;
  float* X = Xs[wv];
  const int tile = blockIdx.x * 4 + wv;
  const int row0 = tile << 4;

  const v16h* F_ge1 = (const v16h*)(wb + PW_GE1);
  const v16h* F_ge2 = (const v16h*)(wb + PW_GE2);
  const v16h* F_vp  = (const v16h*)(wb + PW_VP);
  const v16h* F_qw  = (const v16h*)(wb + PW_QW);
  const v16h* F_kw  = (const v16h*)(wb + PW_KW);
  const v16h* F_bd1 = (const v16h*)(wb + PW_BD1);
  const float* ge1b = (const float*)(wb + PB_GE1B);
  const float* ge2b = (const float*)(wb + PB_GE2B);
  const float* vpb  = (const float*)(wb + PB_VPB);
  const float* qb   = (const float*)(wb + PB_QB);
  const float* kbv  = (const float*)(wb + PB_KB);
  const float* b1b  = (const float*)(wb + PB_BD1B);
  const float* b2w  = (const float*)(wb + PB_BD2W);
  const float  b2b  = *(const float*)(wb + PB_BD2B);

  const int chl = lane & 15;
  const int mh  = (lane >> 4) << 3;

  // stage jafar tile into LDS, zero-padded to 32 columns (unguarded A builds after this)
  if (lane < 16) {
    const float* src = jf + (size_t)(row0 + lane) * GEO;
    float* dr = X + lane * 64;
#pragma unroll
    for (int k = 0; k < GEO; ++k) dr[k] = src[k];
#pragma unroll
    for (int k = GEO; k < 32; ++k) dr[k] = 0.0f;
  }
  v16h aj = a_frag_rows(X, 64, lane, 0);

  // geom layer 1: 12 -> 64 (bn+relu folded)
#pragma unroll
  for (int ct = 0; ct < 4; ++ct) {
    v8f c = {};
    v16h bw = F_ge1[ct * 32 + lane];
    WMMA_F16(c, aj, bw);
    const int ch = (ct << 4) + chl;
    const float bb = ge1b[ch];
#pragma unroll
    for (int r = 0; r < 8; ++r) {
      float v = c[r] + bb; v = v > 0.0f ? v : 0.0f;
      X[(r + mh) * 64 + ch] = v;
    }
  }

  // geom layer 2: 64 -> 64 (bn+relu folded)
  v16h g0 = a_frag_rows(X, 64, lane, 0);
  v16h g1 = a_frag_rows(X, 64, lane, 32);
#pragma unroll
  for (int ct = 0; ct < 4; ++ct) {
    v8f c = {};
    WMMA_F16(c, g0, F_ge2[(ct * 2 + 0) * 32 + lane]);
    WMMA_F16(c, g1, F_ge2[(ct * 2 + 1) * 32 + lane]);
    const int ch = (ct << 4) + chl;
    const float bb = ge2b[ch];
#pragma unroll
    for (int r = 0; r < 8; ++r) {
      float v = c[r] + bb; v = v > 0.0f ? v : 0.0f;
      X[(r + mh) * 64 + ch] = v;                 // geom_emb
    }
  }

  // geom_emb A fragments kept in registers, reused by Q, K, boundary head
  v16h e0 = a_frag_rows(X, 64, lane, 0);
  v16h e1 = a_frag_rows(X, 64, lane, 32);

  // Q projection
#pragma unroll
  for (int ct = 0; ct < 4; ++ct) {
    v8f c = {};
    WMMA_F16(c, e0, F_qw[(ct * 2 + 0) * 32 + lane]);
    WMMA_F16(c, e1, F_qw[(ct * 2 + 1) * 32 + lane]);
    const int ch = (ct << 4) + chl;
    const float bb = qb[ch];
#pragma unroll
    for (int r = 0; r < 8; ++r) Qo[(size_t)(row0 + r + mh) * 64 + ch] = c[r] + bb;
  }
  // K projection
#pragma unroll
  for (int ct = 0; ct < 4; ++ct) {
    v8f c = {};
    WMMA_F16(c, e0, F_kw[(ct * 2 + 0) * 32 + lane]);
    WMMA_F16(c, e1, F_kw[(ct * 2 + 1) * 32 + lane]);
    const int ch = (ct << 4) + chl;
    const float bb = kbv[ch];
#pragma unroll
    for (int r = 0; r < 8; ++r) Ko[(size_t)(row0 + r + mh) * 64 + ch] = c[r] + bb;
  }
  // boundary head: 64 -> 32 (bn+relu) -> 1
#pragma unroll
  for (int ct = 0; ct < 2; ++ct) {
    v8f c = {};
    WMMA_F16(c, e0, F_bd1[(ct * 2 + 0) * 32 + lane]);
    WMMA_F16(c, e1, F_bd1[(ct * 2 + 1) * 32 + lane]);
    const int ch = (ct << 4) + chl;               // 0..31
    const float bb = b1b[ch];
#pragma unroll
    for (int r = 0; r < 8; ++r) {
      float v = c[r] + bb; v = v > 0.0f ? v : 0.0f;
      X[(r + mh) * 64 + ch] = v;                  // h in X[:, 0:32]
    }
  }
  if (lane < 16) {
    float acc = b2b;
#pragma unroll
    for (int k = 0; k < 32; ++k) acc += b2w[k] * X[lane * 64 + k];
    bdy_out[row0 + lane] = acc;
  }

  // V projection: 192 -> 64 (bn+relu folded); all K indices provably < 192 -> unguarded
  v16h av[6];
#pragma unroll
  for (int kt = 0; kt < 6; ++kt)
    av[kt] = a_frag_rows(sem + (size_t)row0 * SEMD, SEMD, lane, kt * 32);
#pragma unroll
  for (int ct = 0; ct < 4; ++ct) {
    v8f c = {};
#pragma unroll
    for (int kt = 0; kt < 6; ++kt) WMMA_F16(c, av[kt], F_vp[(ct * 6 + kt) * 32 + lane]);
    const int ch = (ct << 4) + chl;
    const float bb = vpb[ch];
#pragma unroll
    for (int r = 0; r < 8; ++r) {
      float v = c[r] + bb; v = v > 0.0f ? v : 0.0f;
      Vo[(size_t)(row0 + r + mh) * 64 + ch] = v;
    }
  }
}

// ---------------- kernel 3: attention + classifier (one wave per point) ----------------
__global__ __launch_bounds__(128) void attn_kernel(
    const float* __restrict__ xyz, const int* __restrict__ kidx,
    const unsigned char* __restrict__ wb,
    const float* __restrict__ Qf, const float* __restrict__ Kf, const float* __restrict__ Vf,
    const float* __restrict__ clsw, const float* __restrict__ clsb,
    float* __restrict__ out) {
  __shared__ float PEs[4][16 * 64];
  __shared__ float Qs_[4][64];
  __shared__ float Rs_[4][64];
  __shared__ float Ls_[4][16];
  __shared__ float As_[4][16];
  __shared__ int   Js_[4][16];
  const int lane = threadIdx.x & 31;
  const int wv   = threadIdx.x >> 5;
  float* PE = PEs[wv]; float* Qs = Qs_[wv]; float* Rs = Rs_[wv];
  float* Ls = Ls_[wv]; float* As = As_[wv]; int* Js = Js_[wv];

  const int id   = blockIdx.x * 4 + wv;     // 0..BN-1
  const int b    = id >> 13;
  const int n    = id & (NPTS - 1);
  const int base = b << 13;

  const v16h* F_rp1 = (const v16h*)(wb + PW_RP1);
  const v16h* F_rp2 = (const v16h*)(wb + PW_RP2);
  const float* r1b = (const float*)(wb + PB_RP1B);
  const float* r2b = (const float*)(wb + PB_RP2B);

  const int chl = lane & 15;
  const int mh  = (lane >> 4) << 3;

  // stage Q row
  Qs[lane]      = Qf[(size_t)id * 64 + lane];
  Qs[lane + 32] = Qf[(size_t)id * 64 + lane + 32];

  // neighbor indices + relative positions (lane m holds A-fragment row m; rest zero-padded)
  v16h ar = {};
  if (lane < 16) {
    const int j = kidx[id * KNNK + lane];
    Js[lane] = j;
    const float* pq = xyz + (size_t)(base + n) * 3;
    const float* pj = xyz + (size_t)(base + j) * 3;
    ar[0] = (_Float16)(pq[0] - pj[0]);
    ar[1] = (_Float16)(pq[1] - pj[1]);
    ar[2] = (_Float16)(pq[2] - pj[2]);
  }

  // pos-enc layer 1: 3 -> 64 (bn+relu folded; zero-padding baked into fragments)
#pragma unroll
  for (int ct = 0; ct < 4; ++ct) {
    v8f c = {};
    WMMA_F16(c, ar, F_rp1[ct * 32 + lane]);
    const int ch = (ct << 4) + chl;
    const float bb = r1b[ch];
#pragma unroll
    for (int r = 0; r < 8; ++r) {
      float v = c[r] + bb; v = v > 0.0f ? v : 0.0f;
      PE[(r + mh) * 64 + ch] = v;
    }
  }

  // pos-enc layer 2 (bias only) + gathered-K add -> S (overwrites PE)
  v16h p0 = a_frag_rows(PE, 64, lane, 0);
  v16h p1 = a_frag_rows(PE, 64, lane, 32);
#pragma unroll
  for (int ct = 0; ct < 4; ++ct) {
    v8f c = {};
    WMMA_F16(c, p0, F_rp2[(ct * 2 + 0) * 32 + lane]);
    WMMA_F16(c, p1, F_rp2[(ct * 2 + 1) * 32 + lane]);
    const int ch = (ct << 4) + chl;
    const float bb = r2b[ch];
#pragma unroll
    for (int r = 0; r < 8; ++r) {
      const int m = r + mh;
      const int jm = Js[m];
      PE[m * 64 + ch] = c[r] + bb + Kf[(size_t)(base + jm) * 64 + ch];
    }
  }

  // attention logits and softmax over 16 neighbors
  if (lane < 16) {
    float acc = 0.0f;
#pragma unroll
    for (int cc = 0; cc < 64; ++cc) acc += Qs[cc] * PE[lane * 64 + cc];
    Ls[lane] = acc * 0.125f;                    // 1/sqrt(64)
  }
  if (lane < 16) {
    float mx = Ls[0];
#pragma unroll
    for (int i = 1; i < 16; ++i) mx = fmaxf(mx, Ls[i]);
    float s = 0.0f, mye = 0.0f;
#pragma unroll
    for (int i = 0; i < 16; ++i) {
      const float e = __expf(Ls[i] - mx);
      s += e;
      if (i == lane) mye = e;
    }
    const float a = mye / s;
    As[lane] = a;
    out[O_AFF + (size_t)id * KNNK + lane] = a;
  }

  // refined = sum_m a[m]*V[j_m] + V[n]; 2 channels per lane
  {
    const int c0 = lane, c1 = lane + 32;
    float acc0 = Vf[(size_t)id * 64 + c0];
    float acc1 = Vf[(size_t)id * 64 + c1];
#pragma unroll
    for (int m = 0; m < 16; ++m) {
      const int jm = Js[m];
      const float am = As[m];
      const float* vr = Vf + (size_t)(base + jm) * 64;
      acc0 += am * vr[c0];
      acc1 += am * vr[c1];
    }
    out[O_REF + (size_t)id * 64 + c0] = acc0;
    out[O_REF + (size_t)id * 64 + c1] = acc1;
    Rs[c0] = acc0;
    Rs[c1] = acc1;
  }

  // classifier 64 -> 13
  if (lane < NCLS) {
    float acc = clsb[lane];
#pragma unroll
    for (int cc = 0; cc < 64; ++cc) acc += clsw[lane * 64 + cc] * Rs[cc];
    out[O_LOGITS + (size_t)id * NCLS + lane] = acc;
  }
}

// ---------------- host entry ----------------
extern "C" void kernel_launch(void* const* d_in, const int* in_sizes, int n_in,
                              void* d_out, int out_size, void* d_ws, size_t ws_size,
                              hipStream_t stream) {
  (void)in_sizes; (void)n_in; (void)out_size; (void)ws_size;
  const float* xyz = (const float*)d_in[0];
  const float* jf  = (const float*)d_in[1];
  const float* sem = (const float*)d_in[2];
#define PF(i) ((const float*)d_in[(i)])

  unsigned char* ws = (unsigned char*)d_ws;
  int*   kidx = (int*)(ws + WS_KIDX);
  float* Qf   = (float*)(ws + WS_Q);
  float* Kf   = (float*)(ws + WS_K);
  float* Vf   = (float*)(ws + WS_V);
  unsigned char* wb = ws + WS_PARAM;
  float* out = (float*)d_out;

  prep_kernel<<<1, 256, 0, stream>>>(
      PF(3), PF(4), PF(5), PF(6),      // ge_w1,b1,g1,be1
      PF(7), PF(8), PF(9), PF(10),     // ge_w2,b2,g2,be2
      PF(11), PF(12), PF(13), PF(14),  // vp
      PF(15), PF(16), PF(17), PF(18),  // q,k
      PF(19), PF(20), PF(21), PF(22),  // rp1
      PF(23), PF(24),                  // rp2
      PF(25), PF(26), PF(27), PF(28),  // bd1
      PF(29), PF(30),                  // bd2
      wb);

  knn_kernel<<<BN / 256, 256, 0, stream>>>(xyz, kidx, out + O_KIDX);

  feat_kernel<<<BN / 64, 128, 0, stream>>>(jf, sem, wb, Qf, Kf, Vf, out + O_BDY);

  attn_kernel<<<BN / 4, 128, 0, stream>>>(xyz, kidx, wb, Qf, Kf, Vf,
                                          PF(31), PF(32), out);
#undef PF
}
